// PFRNPolicy_2104533975421
// MI455X (gfx1250) — matrix operand order repaired and checked by reading, not compiled, not run on hardware
//
#include <hip/hip_runtime.h>
#include <hip/hip_bf16.h>

// ---------------------------------------------------------------------------
// Fused policy-network forward for MI455X (gfx1250).
//   d_in[0]  = obs   (int32, B x 273)
//   d_in[1..33] = params in dict insertion order (see enum below)
//   d_out = [logits (B x 8) | value (B)]  float32
//   d_ws  = f16 transposed-weight staging (~484 KB)
// All GEMMs use v_wmma_f32_16x16x32_f16 on 16-row batch tiles.
// ---------------------------------------------------------------------------

typedef __attribute__((ext_vector_type(16))) _Float16 v16h;
typedef __attribute__((ext_vector_type(8)))  _Float16 v8h;
typedef __attribute__((ext_vector_type(8)))  float    v8f;

// f16 weight-buffer offsets (in halfs), all 32B-aligned in bytes.
#define OFF_W1S 0        // scalar_w1^T  [256][320]
#define OFF_W2S 81920    // scalar_w2^T  [128][256]
#define OFF_WN1 114688   // npc_w1^T     [32][32]
#define OFF_WN2 115712   // npc_w2^T     [32][32]
#define OFF_WV1 116736   // visit_w1^T   [64][1024]
#define OFF_WV2 182272   // visit_w2^T   [64][64]
#define OFF_WF  186368   // fusion_w^T   [256][224]
#define OFF_WAV 243712   // [actor|value|pad]^T [16][256]
#define WBUF_HALFS 247808

static __device__ __forceinline__ int imin(int a, int b) { return a < b ? a : b; }

// ---- WMMA tile helpers -----------------------------------------------------
// A: row-major [M][K] in LDS. ISA 16-bit A layout: lane l -> row (l&15),
// chunks of 8 halfs at k0 + 8*(l>=16) and +16.
static __device__ __forceinline__ v16h load_a_tile(const _Float16* A, int lda,
                                                   int k0, int lane) {
  int m  = lane & 15;
  int kh = (lane >> 4) * 8;
  const _Float16* p = A + m * lda + k0 + kh;
  union { v16h v; v8h h[2]; } u;
  u.h[0] = *(const v8h*)(p);
  u.h[1] = *(const v8h*)(p + 16);
  return u.v;
}

// B stored transposed [N][K]: lane l -> column (l&15), chunks of 8 halfs at
// k0 + 8*(l>=16) and +16 (same k order as A).
static __device__ __forceinline__ v16h load_b_tile(const _Float16* B, int ldb,
                                                   int k0, int lane) {
  int n  = lane & 15;
  int kh = (lane >> 4) * 8;
  const _Float16* p = B + n * ldb + k0 + kh;
  union { v16h v; v8h h[2]; } u;
  u.h[0] = *(const v8h*)(p);
  u.h[1] = *(const v8h*)(p + 16);
  return u.v;
}

static __device__ __forceinline__ v8f gemm_tile(const _Float16* A, int lda,
                                                const _Float16* Bt, int ldb,
                                                int ktiles, int lane) {
  v8f acc = {};
  for (int kt = 0; kt < ktiles; ++kt) {
    v16h a = load_a_tile(A, lda, kt * 32, lane);
    v16h b = load_b_tile(Bt, ldb, kt * 32, lane);
    acc = __builtin_amdgcn_wmma_f32_16x16x32_f16(false, a, false, b,
                                                 (short)0, acc, false, false);
  }
  return acc;
}

// D layout: VGPR r, lane l -> (M = r + 8*(l>=16), N = l&15).
static __device__ __forceinline__ void store_relu_f16(v8f acc, _Float16* Dt,
                                                      int ldd, const float* bias,
                                                      int lane) {
  int n = lane & 15, mb = (lane >> 4) * 8;
  float b = bias[n];
  for (int r = 0; r < 8; ++r) {
    float v = acc[r] + b;
    Dt[(mb + r) * ldd + n] = (_Float16)(v > 0.f ? v : 0.f);
  }
}

static __device__ __forceinline__ void store_f32(v8f acc, float* Dt, int ldd,
                                                 const float* bias, int lane) {
  int n = lane & 15, mb = (lane >> 4) * 8;
  float b = bias[n];
  for (int r = 0; r < 8; ++r)
    Dt[(mb + r) * ldd + n] = acc[r] + b;
}

// ---- weight prep -----------------------------------------------------------
__global__ void prep_transpose(const float* __restrict__ src,
                               _Float16* __restrict__ dst,
                               int N, int Ksrc, int Kpad) {
  int i = blockIdx.x * blockDim.x + threadIdx.x;
  if (i >= N * Kpad) return;
  int n = i / Kpad, k = i - n * Kpad;
  dst[i] = (k < Ksrc) ? (_Float16)src[k * N + n] : (_Float16)0.f;
}

__global__ void prep_av(const float* __restrict__ aw, const float* __restrict__ vw,
                        _Float16* __restrict__ dst) {
  int i = blockIdx.x * blockDim.x + threadIdx.x;
  if (i >= 16 * 256) return;
  int n = i >> 8, k = i & 255;
  float v = (n < 8) ? aw[k * 8 + n] : (n == 8 ? vw[k] : 0.f);
  dst[i] = (_Float16)v;
}

// ---- fused forward ---------------------------------------------------------
struct KParams {
  const int* obs;
  const float *sp_emb, *mg_emb, *mn_emb, *ml_emb, *dir_emb, *run_emb, *tr_emb,
              *bo_emb, *we_emb, *ty_emb, *ngfx_emb, *ndir_emb, *nmove_emb;
  const float *sb1, *sb2, *nb1, *nb2, *vb1, *vb2, *fb, *lnw, *lnb, *ab, *vbias;
  const _Float16* w;
  float* out;
  int nrows;
};

__global__ __launch_bounds__(256) void policy_fused(KParams P) {
  // 56 KB hand-overlapped LDS (regions fenced by __syncthreads between stages)
  __shared__ __align__(64) char smem[57344];
  _Float16* FUSED = (_Float16*)(smem + 0);      // 16x224, persists to the end
  float*    ACT   = (float*)   (smem + 7168);   // 240 npc-active floats
  _Float16* H1    = (_Float16*)(smem + 8192);   // 16x256 scalar hidden
  _Float16* ASC   = (_Float16*)(smem + 16384);  // 16x320 scalar features
  _Float16* NPCIN = (_Float16*)(smem + 26624);  // 240x32 npc features
  _Float16* NH    = (_Float16*)(smem + 41984);  // 240x32 npc hidden
  _Float16* NF    = (_Float16*)(smem + 8192);   // 240x32 (aliases H1/ASC, later)
  _Float16* VIS   = (_Float16*)(smem + 8192);   // 16x1024 visit bits (later)
  _Float16* V1    = (_Float16*)(smem + 40960);  // 16x64 visit hidden
  float*    F     = (float*)   (smem + 8192);   // 16x256 pre-LN (later)
  _Float16* HID   = (_Float16*)(smem + 24576);  // 16x256 post-LN/ReLU

  const int tid = threadIdx.x, wave = tid >> 5, lane = tid & 31;
  const int row0 = blockIdx.x * 16;

  const _Float16* W1S = P.w + OFF_W1S;
  const _Float16* W2S = P.w + OFF_W2S;
  const _Float16* WN1 = P.w + OFF_WN1;
  const _Float16* WN2 = P.w + OFF_WN2;
  const _Float16* WV1 = P.w + OFF_WV1;
  const _Float16* WV2 = P.w + OFF_WV2;
  const _Float16* WF  = P.w + OFF_WF;
  const _Float16* WAV = P.w + OFF_WAV;

  // ---- Phase A: build scalar + npc features --------------------------------
  __builtin_prefetch(P.obs + (row0 + (tid >> 4)) * 273 + 145, 0, 0);
  if (tid < 16) {
    const int r = tid;
    const int* o = P.obs + (long)(row0 + r) * 273;
    _Float16* d = ASC + r * 320;
    int k = 0;
    int raw = o[0] | (o[1] << 8);
    d[k++] = (_Float16)((raw >= 32768 ? raw - 65536 : raw) * (1.f / 256.f));
    raw = o[2] | (o[3] << 8);
    d[k++] = (_Float16)((raw >= 32768 ? raw - 65536 : raw) * (1.f / 256.f));
    { const float* e = P.mg_emb + imin(o[4], 63) * 16;  for (int j = 0; j < 16; ++j) d[k++] = (_Float16)e[j]; }
    { const float* e = P.mn_emb + imin(o[5], 255) * 16; for (int j = 0; j < 16; ++j) d[k++] = (_Float16)e[j]; }
    { const float* e = P.ml_emb + imin(o[6], 255) * 16; for (int j = 0; j < 16; ++j) d[k++] = (_Float16)e[j]; }
    { const float* e = P.dir_emb + imin(o[7], 7) * 8;   for (int j = 0; j < 8;  ++j) d[k++] = (_Float16)e[j]; }
    { int v = o[8];  for (int j = 0; j < 8; ++j) d[k++] = (_Float16)(float)((v >> j) & 1); }
    { const float* e = P.run_emb + imin(o[9], 7) * 4;   for (int j = 0; j < 4;  ++j) d[k++] = (_Float16)e[j]; }
    { const float* e = P.tr_emb + imin(o[10], 7) * 4;   for (int j = 0; j < 4;  ++j) d[k++] = (_Float16)e[j]; }
    d[k++] = (_Float16)(float)imin(o[11], 1);
    { const float* e = P.bo_emb + imin(o[12], 7) * 4;   for (int j = 0; j < 4;  ++j) d[k++] = (_Float16)e[j]; }
    for (int p6 = 0; p6 < 6; ++p6) {
      const int* q = o + 13 + p6 * 6;
      int sp = imin(q[0] | (q[1] << 8), 512);
      const float* e = P.sp_emb + sp * 16;
      for (int j = 0; j < 16; ++j) d[k++] = (_Float16)e[j];
      d[k++] = (_Float16)(q[2] * (1.f / 100.f));
      d[k++] = (_Float16)(q[3] * (1.f / 255.f));
      { int v = q[4]; for (int j = 0; j < 8; ++j) d[k++] = (_Float16)(float)((v >> j) & 1); }
      { const float* t = P.ty_emb + imin(q[5], 31) * 8; for (int j = 0; j < 8; ++j) d[k++] = (_Float16)t[j]; }
    }
    { int v = o[49]; for (int j = 0; j < 8; ++j) d[k++] = (_Float16)(float)((v >> j) & 1); }
    d[k++] = (_Float16)((o[50] | (o[51] << 8)) * (1.f / 65535.f));
    { const float* e = P.we_emb + imin(o[52], 15) * 8;  for (int j = 0; j < 8; ++j) d[k++] = (_Float16)e[j]; }
    d[k++] = (_Float16)((o[53] | (o[54] << 8)) * (1.f / 24576.f));
    for (; k < 320; ++k) d[k] = (_Float16)0.f;
  } else {
    const int pair = tid - 16;                  // 240 (row, npc) pairs
    const int r = pair / 15, i = pair - r * 15;
    const int* q = P.obs + (long)(row0 + r) * 273 + 55 + i * 6;
    _Float16* d = NPCIN + pair * 32;
    int v0 = q[0], v1 = q[1];
    d[0] = (_Float16)((v0 >= 128 ? v0 - 256 : v0) * (1.f / 127.f));
    d[1] = (_Float16)((v1 >= 128 ? v1 - 256 : v1) * (1.f / 127.f));
    { const float* g = P.ngfx_emb + imin(q[2], 255) * 16; for (int j = 0; j < 16; ++j) d[2 + j] = (_Float16)g[j]; }
    { const float* g = P.ndir_emb + imin(q[3], 7) * 8;    for (int j = 0; j < 8;  ++j) d[18 + j] = (_Float16)g[j]; }
    float a = (float)q[4];
    d[26] = (_Float16)a;
    ACT[pair] = a;
    { const float* g = P.nmove_emb + imin(q[5], 31) * 4;  for (int j = 0; j < 4;  ++j) d[27 + j] = (_Float16)g[j]; }
    d[31] = (_Float16)0.f;
  }
  __syncthreads();

  // ---- scalar layer 1: 16x320 @ 320x256 -> H1 ------------------------------
  for (int nt = wave; nt < 16; nt += 8) {
    v8f acc = gemm_tile(ASC, 320, W1S + nt * 16 * 320, 320, 10, lane);
    store_relu_f16(acc, H1 + nt * 16, 256, P.sb1 + nt * 16, lane);
  }
  __syncthreads();

  // ---- scalar layer 2: 16x256 @ 256x128 -> FUSED[:,0:128] ------------------
  {
    int nt = wave;
    v8f acc = gemm_tile(H1, 256, W2S + nt * 16 * 256, 256, 8, lane);
    store_relu_f16(acc, FUSED + nt * 16, 224, P.sb2 + nt * 16, lane);
  }
  __syncthreads();

  // ---- npc layer 1: 240x32 @ 32x32 -> NH ----------------------------------
  for (int id = wave; id < 30; id += 8) {
    int mt = id >> 1, nt = id & 1;
    v8f acc = gemm_tile(NPCIN + mt * 16 * 32, 32, WN1 + nt * 16 * 32, 32, 1, lane);
    store_relu_f16(acc, NH + mt * 16 * 32 + nt * 16, 32, P.nb1 + nt * 16, lane);
  }
  __syncthreads();

  // ---- npc layer 2: 240x32 @ 32x32 -> NF (aliases dead H1/ASC) -------------
  for (int id = wave; id < 30; id += 8) {
    int mt = id >> 1, nt = id & 1;
    v8f acc = gemm_tile(NH + mt * 16 * 32, 32, WN2 + nt * 16 * 32, 32, 1, lane);
    store_relu_f16(acc, NF + mt * 16 * 32 + nt * 16, 32, P.nb2 + nt * 16, lane);
  }
  __syncthreads();

  // ---- masked max pool over 15 npcs -> FUSED[:,128:160] --------------------
  for (int it = tid; it < 512; it += 256) {
    int r = it >> 5, c = it & 31;
    float best = -3.4e38f, asum = 0.f;
    for (int i = 0; i < 15; ++i) {
      float a = ACT[r * 15 + i];
      float f = (float)NF[(r * 15 + i) * 32 + c];
      float mv = f * a + (1.f - a) * (-1e9f);
      best = fmaxf(best, mv);
      asum += a;
    }
    FUSED[r * 224 + 128 + c] = (_Float16)(asum > 0.f ? best : 0.f);
  }
  __syncthreads();

  // ---- build visit bit features (reuses NF/H1/ASC/NPCIN region) ------------
  {
    int r = tid >> 4, g = tid & 15;
    const int* o = P.obs + (long)(row0 + r) * 273 + 145 + g * 8;
    _Float16* d = VIS + r * 1024 + g * 64;
    for (int b = 0; b < 8; ++b) {
      int v = o[b];
      for (int i = 0; i < 8; ++i) d[b * 8 + i] = (_Float16)(float)((v >> i) & 1);
    }
  }
  __syncthreads();

  // ---- visit layer 1: 16x1024 @ 1024x64 -> V1 (waves 0..3) -----------------
  if (wave < 4) {
    v8f acc = gemm_tile(VIS, 1024, WV1 + wave * 16 * 1024, 1024, 32, lane);
    store_relu_f16(acc, V1 + wave * 16, 64, P.vb1 + wave * 16, lane);
  }
  __syncthreads();

  // ---- visit layer 2: 16x64 @ 64x64 -> FUSED[:,160:224] --------------------
  if (wave < 4) {
    v8f acc = gemm_tile(V1, 64, WV2 + wave * 16 * 64, 64, 2, lane);
    store_relu_f16(acc, FUSED + 160 + wave * 16, 224, P.vb2 + wave * 16, lane);
  }
  __syncthreads();

  // ---- fusion: 16x224 @ 224x256 -> F (f32, no relu) ------------------------
  for (int nt = wave; nt < 16; nt += 8) {
    v8f acc = gemm_tile(FUSED, 224, WF + nt * 16 * 224, 224, 7, lane);
    store_f32(acc, F + nt * 16, 256, P.fb + nt * 16, lane);
  }
  __syncthreads();

  // ---- LayerNorm + ReLU -> HID --------------------------------------------
  if (tid < 16) {
    const float* fr = F + tid * 256;
    float s = 0.f, s2 = 0.f;
    for (int c = 0; c < 256; ++c) { float v = fr[c]; s += v; s2 += v * v; }
    float mu = s * (1.f / 256.f);
    float var = s2 * (1.f / 256.f) - mu * mu;
    float rstd = rsqrtf((var > 0.f ? var : 0.f) + 1e-5f);
    _Float16* hr = HID + tid * 256;
    for (int c = 0; c < 256; ++c) {
      float v = (fr[c] - mu) * rstd * P.lnw[c] + P.lnb[c];
      hr[c] = (_Float16)(v > 0.f ? v : 0.f);
    }
  }
  __syncthreads();

  // ---- actor + value: 16x256 @ 256x16 (cols: 0-7 actor, 8 value) -----------
  if (wave == 0) {
    v8f acc = gemm_tile(HID, 256, WAV, 256, 8, lane);
    int n = lane & 15, mb = (lane >> 4) * 8;
    for (int r = 0; r < 8; ++r) {
      int row = row0 + mb + r;
      float v = acc[r];
      if (n < 8)        P.out[(long)row * 8 + n] = v + P.ab[n];
      else if (n == 8)  P.out[(long)P.nrows * 8 + row] = v + P.vbias[0];
    }
  }
}

// ---------------------------------------------------------------------------
extern "C" void kernel_launch(void* const* d_in, const int* in_sizes, int n_in,
                              void* d_out, int out_size, void* d_ws, size_t ws_size,
                              hipStream_t stream) {
  // Params in setup_inputs() dict order (obs first, then nested params dict).
  const int*   obs = (const int*)d_in[0];
  const float* sp_emb    = (const float*)d_in[1];
  const float* mg_emb    = (const float*)d_in[2];
  const float* mn_emb    = (const float*)d_in[3];
  const float* ml_emb    = (const float*)d_in[4];
  const float* dir_emb   = (const float*)d_in[5];
  const float* run_emb   = (const float*)d_in[6];
  const float* tr_emb    = (const float*)d_in[7];
  const float* bo_emb    = (const float*)d_in[8];
  const float* we_emb    = (const float*)d_in[9];
  const float* ty_emb    = (const float*)d_in[10];
  const float* ngfx_emb  = (const float*)d_in[11];
  const float* ndir_emb  = (const float*)d_in[12];
  const float* nmove_emb = (const float*)d_in[13];
  const float* scalar_w1 = (const float*)d_in[14];
  const float* scalar_b1 = (const float*)d_in[15];
  const float* scalar_w2 = (const float*)d_in[16];
  const float* scalar_b2 = (const float*)d_in[17];
  const float* npc_w1    = (const float*)d_in[18];
  const float* npc_b1    = (const float*)d_in[19];
  const float* npc_w2    = (const float*)d_in[20];
  const float* npc_b2    = (const float*)d_in[21];
  const float* visit_w1  = (const float*)d_in[22];
  const float* visit_b1  = (const float*)d_in[23];
  const float* visit_w2  = (const float*)d_in[24];
  const float* visit_b2  = (const float*)d_in[25];
  const float* fusion_w  = (const float*)d_in[26];
  const float* fusion_b  = (const float*)d_in[27];
  const float* ln_w      = (const float*)d_in[28];
  const float* ln_b      = (const float*)d_in[29];
  const float* actor_w   = (const float*)d_in[30];
  const float* actor_b   = (const float*)d_in[31];
  const float* value_w   = (const float*)d_in[32];
  const float* value_b   = (const float*)d_in[33];

  _Float16* w = (_Float16*)d_ws;
  const int nrows = in_sizes[0] / 273;

  auto blocks = [](int n) { return (n + 255) / 256; };
  prep_transpose<<<blocks(256 * 320), 256, 0, stream>>>(scalar_w1, w + OFF_W1S, 256, 301, 320);
  prep_transpose<<<blocks(128 * 256), 256, 0, stream>>>(scalar_w2, w + OFF_W2S, 128, 256, 256);
  prep_transpose<<<blocks(32 * 32),   256, 0, stream>>>(npc_w1,    w + OFF_WN1, 32, 31, 32);
  prep_transpose<<<blocks(32 * 32),   256, 0, stream>>>(npc_w2,    w + OFF_WN2, 32, 32, 32);
  prep_transpose<<<blocks(64 * 1024), 256, 0, stream>>>(visit_w1,  w + OFF_WV1, 64, 1024, 1024);
  prep_transpose<<<blocks(64 * 64),   256, 0, stream>>>(visit_w2,  w + OFF_WV2, 64, 64, 64);
  prep_transpose<<<blocks(256 * 224), 256, 0, stream>>>(fusion_w,  w + OFF_WF,  256, 224, 224);
  prep_av<<<blocks(16 * 256), 256, 0, stream>>>(actor_w, value_w, w + OFF_WAV);

  KParams P;
  P.obs = obs;
  P.sp_emb = sp_emb; P.mg_emb = mg_emb; P.mn_emb = mn_emb; P.ml_emb = ml_emb;
  P.dir_emb = dir_emb; P.run_emb = run_emb; P.tr_emb = tr_emb; P.bo_emb = bo_emb;
  P.we_emb = we_emb; P.ty_emb = ty_emb; P.ngfx_emb = ngfx_emb; P.ndir_emb = ndir_emb;
  P.nmove_emb = nmove_emb;
  P.sb1 = scalar_b1; P.sb2 = scalar_b2; P.nb1 = npc_b1; P.nb2 = npc_b2;
  P.vb1 = visit_b1; P.vb2 = visit_b2; P.fb = fusion_b; P.lnw = ln_w; P.lnb = ln_b;
  P.ab = actor_b; P.vbias = value_b;
  P.w = w; P.out = (float*)d_out; P.nrows = nrows;

  policy_fused<<<nrows / 16, 256, 0, stream>>>(P);
}
